// LSTM_87256555586269
// MI455X (gfx1250) — matrix operand order repaired and checked by reading, not compiled, not run on hardware
//
#include <hip/hip_runtime.h>
#include <hip/hip_bf16.h>

// Dims (match reference)
#define T_LEN 1024
#define B_SZ  64
#define D_IN  1024
#define H_SZ  30
#define G4    120   // 4*H
#define NPAD  128   // padded 4H

typedef __attribute__((ext_vector_type(16))) _Float16 v16h;
typedef __attribute__((ext_vector_type(8)))  _Float16 v8h;
typedef __attribute__((ext_vector_type(2)))  _Float16 v2h;
typedef __attribute__((ext_vector_type(8)))  float    v8f;
typedef __attribute__((ext_vector_type(4)))  float    v4f;
typedef __attribute__((ext_vector_type(2)))  float    v2f;

// Hardware tanh if available (CDNA5 has V_TANH_F32); exp fallback otherwise.
#if __has_builtin(__builtin_amdgcn_tanh_f32)
#define TANHF(x) __builtin_amdgcn_tanh_f32(x)
#elif __has_builtin(__builtin_amdgcn_tanhf)
#define TANHF(x) __builtin_amdgcn_tanhf(x)
#else
#define TANHF(x) (2.0f / (1.0f + __expf(-2.0f * (x))) - 1.0f)
#endif
__device__ __forceinline__ float sigf(float x) { return 0.5f + 0.5f * TANHF(0.5f * x); }

#if __has_builtin(__builtin_amdgcn_sched_barrier)
#define SCHED_FENCE() __builtin_amdgcn_sched_barrier(0)
#else
#define SCHED_FENCE()
#endif

// ---------------------------------------------------------------------------
// Pack W_ih0 [120,1024] (f32) into B-fragment-ordered f16:
//   Bp[((kt*8+n)*32+lane)*16 + j],  K = kt*32 + (lane<16?0:8) + (j<8?j:j+8), N = n*16+(lane&15)
// ---------------------------------------------------------------------------
__global__ __launch_bounds__(256) void pack_wih0(const float* __restrict__ W,
                                                 _Float16* __restrict__ Bp) {
    int id = blockIdx.x * 256 + threadIdx.x;         // 0 .. 131071
    int j    =  id        & 15;
    int lane = (id >> 4)  & 31;
    int n    = (id >> 9)  & 7;
    int kt   =  id >> 12;                            // 0..31
    int K = kt * 32 + ((lane < 16) ? 0 : 8) + ((j < 8) ? j : j + 8);    // 0..1023
    int N = n * 16 + (lane & 15);                    // 0..127
    float v = (N < G4) ? W[N * D_IN + K] : 0.0f;
    Bp[id] = (_Float16)v;
}

// Pack the three small [120,30] matrices (W_hh0, W_ih1, W_hh1), K padded 30->32.
__global__ __launch_bounds__(256) void pack_small(const float* __restrict__ W0,
                                                  const float* __restrict__ W1,
                                                  const float* __restrict__ W2,
                                                  _Float16* __restrict__ out) {
    int id = blockIdx.x * 256 + threadIdx.x;         // 0..4095
    int m  = blockIdx.y;                             // 0..2
    const float* W = (m == 0) ? W0 : ((m == 1) ? W1 : W2);
    int j    =  id       & 15;
    int lane = (id >> 4) & 31;
    int n    =  id >> 9;                             // 0..7
    int K = ((lane < 16) ? 0 : 8) + ((j < 8) ? j : j + 8);              // 0..31
    int N = n * 16 + (lane & 15);                    // 0..127
    float v = (N < G4 && K < H_SZ) ? W[N * H_SZ + K] : 0.0f;
    out[m * 4096 + id] = (_Float16)v;
}

// ---------------------------------------------------------------------------
// Big GEMM: xg[row][0..127] = x[row][:] @ W_ih0^T + (b_ih0+b_hh0), rows = T*B.
// sched_barrier-fenced so all 8 B fragments load in one clause (single wait),
// next A-tile x-loads issue before the 8 back-to-back WMMAs.
// ---------------------------------------------------------------------------
__device__ __forceinline__ v16h cvtA(v4f q0, v4f q1, v4f q2, v4f q3) {
    v16h a;
    a[0]  = (_Float16)q0[0];  a[1]  = (_Float16)q0[1];
    a[2]  = (_Float16)q0[2];  a[3]  = (_Float16)q0[3];
    a[4]  = (_Float16)q1[0];  a[5]  = (_Float16)q1[1];
    a[6]  = (_Float16)q1[2];  a[7]  = (_Float16)q1[3];
    a[8]  = (_Float16)q2[0];  a[9]  = (_Float16)q2[1];
    a[10] = (_Float16)q2[2];  a[11] = (_Float16)q2[3];
    a[12] = (_Float16)q3[0];  a[13] = (_Float16)q3[1];
    a[14] = (_Float16)q3[2];  a[15] = (_Float16)q3[3];
    return a;
}

__global__ __launch_bounds__(256) void gemm_xg(const float* __restrict__ x,
                                               const _Float16* __restrict__ Bp,
                                               const float* __restrict__ bih,
                                               const float* __restrict__ bhh,
                                               float* __restrict__ xg) {
    const int lane   = threadIdx.x & 31;
    const int wave   = threadIdx.x >> 5;
    const int lane15 = lane & 15;
    const int hi     = (lane < 16) ? 0 : 8;
    const int rowBase = blockIdx.x * 128 + wave * 16;
    const float* __restrict__ xr = x + (size_t)(rowBase + lane15) * D_IN;

    v8f acc[8] = {};

    // preload A-tile inputs for kt=0
    v4f q0 = *(const v4f*)(xr + hi);
    v4f q1 = *(const v4f*)(xr + hi + 4);
    v4f q2 = *(const v4f*)(xr + hi + 16);
    v4f q3 = *(const v4f*)(xr + hi + 20);

    for (int kt = 0; kt < 32; ++kt) {
        // batch all 8 B fragments: fenced so they issue as one clause
        v16h bf[8];
        #pragma unroll
        for (int n = 0; n < 8; ++n)
            bf[n] = *(const v16h*)(Bp + ((size_t)(kt * 8 + n) * 32 + lane) * 16);
        SCHED_FENCE();

        v16h a = cvtA(q0, q1, q2, q3);

        // issue next-iteration x loads before the WMMA chain
        if (kt + 1 < 32) {
            int koff = (kt + 1) * 32 + hi;
            q0 = *(const v4f*)(xr + koff);
            q1 = *(const v4f*)(xr + koff + 4);
            q2 = *(const v4f*)(xr + koff + 16);
            q3 = *(const v4f*)(xr + koff + 20);
        }
        SCHED_FENCE();

        #pragma unroll
        for (int n = 0; n < 8; ++n)
            acc[n] = __builtin_amdgcn_wmma_f32_16x16x32_f16(
                false, a, false, bf[n], (short)0, acc[n], false, false);
    }

    // bias add + store (C layout: elem r -> M = r+hi, N = lane15 within tile)
    #pragma unroll
    for (int n = 0; n < 8; ++n) {
        int col = n * 16 + lane15;
        float bias = (col < G4) ? (bih[col] + bhh[col]) : 0.0f;
        #pragma unroll
        for (int r = 0; r < 8; ++r) {
            xg[(size_t)(rowBase + hi + r) * NPAD + col] = acc[n][r] + bias;
        }
    }
}

// ---------------------------------------------------------------------------
// Sequential fused recurrence: both LSTM layers + output head.
// One workgroup (8 wave32), 4 barriers per step.  Cell state lives in
// registers (static thread->element mapping), layer-1 bias rides the WMMA
// C operand, xg stream double-buffered via register prefetch.
// ---------------------------------------------------------------------------
__device__ __forceinline__ v16h loadA40(const _Float16* __restrict__ h, int r0, int hi) {
    v8h lo = *(const v8h*)(h + r0 * 40 + hi);
    v8h hb = *(const v8h*)(h + r0 * 40 + hi + 16);
    return __builtin_shufflevector(lo, hb, 0, 1, 2, 3, 4, 5, 6, 7,
                                           8, 9, 10, 11, 12, 13, 14, 15);
}

__global__ __launch_bounds__(256) void lstm_seq(const float* __restrict__ xg,
                                                const _Float16* __restrict__ Wp,
                                                const float* __restrict__ bih1,
                                                const float* __restrict__ bhh1,
                                                const float* __restrict__ wout,
                                                const float* __restrict__ bout,
                                                float* __restrict__ out) {
    __shared__ __align__(16) float    ldsXg[2][B_SZ * NPAD];  // 64 KB double buffer
    __shared__ __align__(16) float    ldsG [B_SZ * NPAD];     // 32 KB gate scratch
    __shared__ __align__(16) _Float16 ldsH0[B_SZ * 40];       // h layer0 (f16, 80B rows)
    __shared__ __align__(16) _Float16 ldsH1[B_SZ * 40];       // h layer1
    __shared__ __align__(16) float    ldsHo[B_SZ * 32];       // h layer1 f32 (for head)
    __shared__ float ldsWo[32];

    const int tid    = threadIdx.x;
    const int lane   = tid & 31;
    const int wave   = tid >> 5;                 // owned N-tile
    const int lane15 = lane & 15;
    const int hi     = (lane < 16) ? 0 : 8;

    // ---- init state ----
    for (int i = tid; i < B_SZ * 40; i += 256) { ldsH0[i] = (_Float16)0.0f; ldsH1[i] = (_Float16)0.0f; }
    if (tid < 32) ldsWo[tid] = (tid < H_SZ) ? wout[tid] : 0.0f;
    for (int i = tid; i < (B_SZ * NPAD) / 4; i += 256)
        ((v4f*)ldsXg[0])[i] = ((const v4f*)xg)[i];

    // per-wave resident weight B-fragments (N-tile = wave)
    const v16h bWhh0 = *(const v16h*)(Wp + 0 * 4096 + ((size_t)(wave * 32 + lane)) * 16);
    const v16h bWih1 = *(const v16h*)(Wp + 1 * 4096 + ((size_t)(wave * 32 + lane)) * 16);
    const v16h bWhh1 = *(const v16h*)(Wp + 2 * 4096 + ((size_t)(wave * 32 + lane)) * 16);
    const float bo = bout[0];

    // layer-1 bias as a broadcast C fragment (col = wave*16 + lane15)
    v8f cBias1;
    {
        int col = wave * 16 + lane15;
        float b1 = (col < G4) ? (bih1[col] + bhh1[col]) : 0.0f;
        #pragma unroll
        for (int r = 0; r < 8; ++r) cBias1[r] = b1;
    }

    // static elementwise mapping: thread handles pairs e = tid + k*256 (960 pairs)
    int offG[4], offH[4], offHo[4];
    bool act[4];
    #pragma unroll
    for (int k = 0; k < 4; ++k) {
        int e = tid + k * 256;
        act[k] = (e < B_SZ * 15);
        int b = e / 15, p = e - b * 15, j2 = 2 * p;
        if (!act[k]) { b = 0; j2 = 0; }
        offG[k]  = b * NPAD + j2;
        offH[k]  = b * 40 + j2;
        offHo[k] = b * 32 + j2;
    }
    v2f cr0[4] = {}, cr1[4] = {};   // cell state in registers, both layers

    __syncthreads();

    v4f pf[8];
    for (int t = 0; t < T_LEN; ++t) {
        // (a) prefetch next xg block to registers
        if (t + 1 < T_LEN) {
            const v4f* src = (const v4f*)(xg + (size_t)(t + 1) * B_SZ * NPAD);
            #pragma unroll
            for (int i = 0; i < 8; ++i) pf[i] = src[tid + i * 256];
        }

        // (b) gates0 = h0 @ Whh0^T
        #pragma unroll
        for (int m = 0; m < 4; ++m) {
            int r0 = m * 16 + lane15;
            v16h a = loadA40(ldsH0, r0, hi);
            v8f c = {};
            c = __builtin_amdgcn_wmma_f32_16x16x32_f16(
                false, a, false, bWhh0, (short)0, c, false, false);
            #pragma unroll
            for (int r = 0; r < 8; ++r)
                ldsG[(m * 16 + hi + r) * NPAD + wave * 16 + lane15] = c[r];
        }
        __syncthreads();

        // (d) layer-0 elementwise (pair-vectorized, c in registers)
        {
            const float* __restrict__ xgc = ldsXg[t & 1];
            #pragma unroll
            for (int k = 0; k < 4; ++k) {
                if (!act[k]) continue;
                const float* g  = ldsG + offG[k];
                const float* xv = xgc  + offG[k];
                v2f gi = *(const v2f*)(g)      + *(const v2f*)(xv);
                v2f gf = *(const v2f*)(g + 30) + *(const v2f*)(xv + 30);
                v2f gg = *(const v2f*)(g + 60) + *(const v2f*)(xv + 60);
                v2f go = *(const v2f*)(g + 90) + *(const v2f*)(xv + 90);
                v2f cn, hn;
                cn[0] = sigf(gf[0]) * cr0[k][0] + sigf(gi[0]) * TANHF(gg[0]);
                cn[1] = sigf(gf[1]) * cr0[k][1] + sigf(gi[1]) * TANHF(gg[1]);
                cr0[k] = cn;
                hn[0] = sigf(go[0]) * TANHF(cn[0]);
                hn[1] = sigf(go[1]) * TANHF(cn[1]);
                v2h hh; hh[0] = (_Float16)hn[0]; hh[1] = (_Float16)hn[1];
                *(v2h*)(ldsH0 + offH[k]) = hh;
            }
        }
        __syncthreads();

        // (e) gates1 = h0_new @ Wih1^T + h1 @ Whh1^T + bias1 (via C operand)
        #pragma unroll
        for (int m = 0; m < 4; ++m) {
            int r0 = m * 16 + lane15;
            v16h a0 = loadA40(ldsH0, r0, hi);
            v16h a1 = loadA40(ldsH1, r0, hi);
            v8f c = cBias1;
            c = __builtin_amdgcn_wmma_f32_16x16x32_f16(
                false, a0, false, bWih1, (short)0, c, false, false);
            c = __builtin_amdgcn_wmma_f32_16x16x32_f16(
                false, a1, false, bWhh1, (short)0, c, false, false);
            #pragma unroll
            for (int r = 0; r < 8; ++r)
                ldsG[(m * 16 + hi + r) * NPAD + wave * 16 + lane15] = c[r];
        }
        __syncthreads();

        // (f) layer-1 elementwise (bias already in gates)
        #pragma unroll
        for (int k = 0; k < 4; ++k) {
            if (!act[k]) continue;
            const float* g = ldsG + offG[k];
            v2f gi = *(const v2f*)(g);
            v2f gf = *(const v2f*)(g + 30);
            v2f gg = *(const v2f*)(g + 60);
            v2f go = *(const v2f*)(g + 90);
            v2f cn, hn;
            cn[0] = sigf(gf[0]) * cr1[k][0] + sigf(gi[0]) * TANHF(gg[0]);
            cn[1] = sigf(gf[1]) * cr1[k][1] + sigf(gi[1]) * TANHF(gg[1]);
            cr1[k] = cn;
            hn[0] = sigf(go[0]) * TANHF(cn[0]);
            hn[1] = sigf(go[1]) * TANHF(cn[1]);
            v2h hh; hh[0] = (_Float16)hn[0]; hh[1] = (_Float16)hn[1];
            *(v2h*)(ldsH1 + offH[k]) = hh;
            *(v2f*)(ldsHo + offHo[k]) = hn;
        }
        __syncthreads();

        // (g) output head (threads 0-63) + xg commit (all threads); no extra
        // barrier needed: next-step ordering is covered by the after-(b) barrier.
        if (tid < B_SZ) {
            const float* hr = ldsHo + tid * 32;
            v4f w0 = *(const v4f*)(ldsWo);      v4f h0v = *(const v4f*)(hr);
            v4f w1 = *(const v4f*)(ldsWo + 4);  v4f h1v = *(const v4f*)(hr + 4);
            v4f w2 = *(const v4f*)(ldsWo + 8);  v4f h2v = *(const v4f*)(hr + 8);
            v4f w3 = *(const v4f*)(ldsWo + 12); v4f h3v = *(const v4f*)(hr + 12);
            v4f w4 = *(const v4f*)(ldsWo + 16); v4f h4v = *(const v4f*)(hr + 16);
            v4f w5 = *(const v4f*)(ldsWo + 20); v4f h5v = *(const v4f*)(hr + 20);
            v4f w6 = *(const v4f*)(ldsWo + 24); v4f h6v = *(const v4f*)(hr + 24);
            v2f w7 = *(const v2f*)(ldsWo + 28); v2f h7v = *(const v2f*)(hr + 28);
            v4f s4 = w0 * h0v + w1 * h1v + w2 * h2v + w3 * h3v + w4 * h4v
                   + w5 * h5v + w6 * h6v;
            float s = s4[0] + s4[1] + s4[2] + s4[3] + w7[0] * h7v[0] + w7[1] * h7v[1];
            out[t * B_SZ + tid] = s + bo;
        }
        if (t + 1 < T_LEN) {
            v4f* dst = (v4f*)ldsXg[(t + 1) & 1];
            #pragma unroll
            for (int i = 0; i < 8; ++i) dst[tid + i * 256] = pf[i];
        }
    }
}

// ---------------------------------------------------------------------------
// Host-side launch
// ---------------------------------------------------------------------------
extern "C" void kernel_launch(void* const* d_in, const int* in_sizes, int n_in,
                              void* d_out, int out_size, void* d_ws, size_t ws_size,
                              hipStream_t stream) {
    const float* x     = (const float*)d_in[0];
    const float* W_ih0 = (const float*)d_in[1];
    const float* W_hh0 = (const float*)d_in[2];
    const float* b_ih0 = (const float*)d_in[3];
    const float* b_hh0 = (const float*)d_in[4];
    const float* W_ih1 = (const float*)d_in[5];
    const float* W_hh1 = (const float*)d_in[6];
    const float* b_ih1 = (const float*)d_in[7];
    const float* b_hh1 = (const float*)d_in[8];
    const float* W_out = (const float*)d_in[9];
    const float* b_out = (const float*)d_in[10];
    float* out = (float*)d_out;

    // workspace carve-up
    char* ws = (char*)d_ws;
    _Float16* Bp = (_Float16*)ws;                      // 262144 B: packed W_ih0
    _Float16* Wp = (_Float16*)(ws + 262144);           //  24576 B: packed Whh0/Wih1/Whh1
    float*    xg = (float*)(ws + 262144 + 24576);      // 33554432 B: xg0 [T*B][128]

    pack_wih0<<<512, 256, 0, stream>>>(W_ih0, Bp);
    pack_small<<<dim3(16, 3), 256, 0, stream>>>(W_hh0, W_ih1, W_hh1, Wp);
    gemm_xg<<<512, 256, 0, stream>>>(x, Bp, b_ih0, b_hh0, xg);
    lstm_seq<<<1, 256, 0, stream>>>(xg, Wp, b_ih1, b_hh1, W_out, b_out, out);
}
